// MLPTexture3D_5222680232372
// MI455X (gfx1250) — matrix operand order repaired and checked by reading, not compiled, not run on hardware
//
#include <hip/hip_runtime.h>
#include <math.h>

typedef __attribute__((ext_vector_type(16))) _Float16 v16h;
typedef __attribute__((ext_vector_type(8)))  float    v8f;

#define HG_LEVELS 16
#define HG_T      (1u << 19)

union AFrag { v16h h; unsigned u[8]; };

__device__ __forceinline__ v8f wmma_f16(v16h a, v16h b, v8f c) {
  // 8 args: (neg_a, A, neg_b, B, c_mod, C, reuse_a, reuse_b)
  return __builtin_amdgcn_wmma_f32_16x16x32_f16(false, a, false, b, (short)0, c,
                                                false, false);
}

__global__ __launch_bounds__(256) void mlptex3d_fused(
    const float* __restrict__ texc,
    const float* __restrict__ table,
    const float* __restrict__ W1,
    const float* __restrict__ W2,
    const float* __restrict__ W3,
    const float* __restrict__ aabb,
    const float* __restrict__ minmax,
    float* __restrict__ out,
    int npts)
{
  __shared__ unsigned encPack[256][HG_LEVELS];        // 16 KB: f16x2 per level
  __shared__ __align__(16) _Float16 hbuf[8][16][32];  // 8 KB: per-wave hidden bounce

  const int tid  = threadIdx.x;
  const int lane = tid & 31;
  const int wave = tid >> 5;
  const int blockBase = blockIdx.x * 256;

  // ===================== Phase 1: hash-grid encode =====================
  {
    int p  = blockBase + tid;
    int pc = p < npts ? p : (npts - 1);          // clamp reads, keep EXEC full
    float tx = texc[pc * 3 + 0];
    float ty = texc[pc * 3 + 1];
    float tz = texc[pc * 3 + 2];
    float lx = aabb[0], ly = aabb[1], lz = aabb[2];
    float hx = aabb[3], hy = aabb[4], hz = aabb[5];
    float x0 = fminf(fmaxf((tx - lx) / (hx - lx), 0.f), 1.f);
    float y0 = fminf(fmaxf((ty - ly) / (hy - ly), 0.f), 1.f);
    float z0 = fminf(fmaxf((tz - lz) / (hz - lz), 0.f), 1.f);

#pragma unroll 1
    for (int l = 0; l < HG_LEVELS; ++l) {
      // scale_l = 16 * 256^(l/15) - 1 = 2^(4 + 8l/15) - 1
      float scale = exp2f((float)l * (8.0f / 15.0f)) * 16.0f - 1.0f;
      int   res   = (int)ceilf(scale) + 1;
      bool  dense = ((long long)res * res * res) <= (long long)HG_T;

      float px = x0 * scale + 0.5f;
      float py = y0 * scale + 0.5f;
      float pz = z0 * scale + 0.5f;
      float fx = floorf(px), fy = floorf(py), fz = floorf(pz);
      float rx = px - fx, ry = py - fy, rz = pz - fz;
      int ix = (int)fx, iy = (int)fy, iz = (int)fz;

      const float* lt = table + (size_t)l * (size_t)HG_T * 2u;
      float f0 = 0.f, f1 = 0.f;
#pragma unroll
      for (int c = 0; c < 8; ++c) {
        int ox = c & 1, oy = (c >> 1) & 1, oz = (c >> 2) & 1;
        unsigned idx;
        if (dense) {
          int cx = min(ix + ox, res - 1);
          int cy = min(iy + oy, res - 1);
          int cz = min(iz + oz, res - 1);
          idx = (unsigned)(cx + cy * res + cz * res * res);
        } else {
          unsigned cx = (unsigned)(ix + ox);
          unsigned cy = (unsigned)(iy + oy);
          unsigned cz = (unsigned)(iz + oz);
          idx = (cx * 1u) ^ (cy * 2654435761u) ^ (cz * 805459861u);
          idx &= (HG_T - 1u);
        }
        float w = (ox ? rx : 1.f - rx) * (oy ? ry : 1.f - ry) *
                  (oz ? rz : 1.f - rz);
        float2 tv = *reinterpret_cast<const float2*>(lt + (size_t)idx * 2u);
        f0 = fmaf(w, tv.x, f0);
        f1 = fmaf(w, tv.y, f1);
      }
      union { _Float16 f[2]; unsigned u; } pk;
      pk.f[0] = (_Float16)f0;
      pk.f[1] = (_Float16)f1;
      encPack[tid][l] = pk.u;   // packed pair = features K=2l, 2l+1
    }
  }
  __syncthreads();

  // ===================== Phase 2: fused MLP via WMMA =====================
  const bool hi   = lane >= 16;
  const int  ncol = lane & 15;
  const int  koff = hi ? 16 : 0;   // B-frag K offset: contiguous K = 16h + s
  const int  mrow = hi ? 8 : 0;    // C/D-frag M offset per ISA layout

  // B fragments: lane holds column N = ncol(+Nbase), halves = 16 consecutive K
  v16h b1lo, b1hi, b2lo, b2hi, b3;
  {
    const float* r;
    r = W1 + (size_t)ncol * 32 + koff;
#pragma unroll
    for (int i = 0; i < 16; ++i) b1lo[i] = (_Float16)r[i];
    r = W1 + (size_t)(ncol + 16) * 32 + koff;
#pragma unroll
    for (int i = 0; i < 16; ++i) b1hi[i] = (_Float16)r[i];
    r = W2 + (size_t)ncol * 32 + koff;
#pragma unroll
    for (int i = 0; i < 16; ++i) b2lo[i] = (_Float16)r[i];
    r = W2 + (size_t)(ncol + 16) * 32 + koff;
#pragma unroll
    for (int i = 0; i < 16; ++i) b2hi[i] = (_Float16)r[i];
    if (ncol < 3) {
      r = W3 + (size_t)ncol * 32 + koff;
#pragma unroll
      for (int i = 0; i < 16; ++i) b3[i] = (_Float16)r[i];
    } else {
#pragma unroll
      for (int i = 0; i < 16; ++i) b3[i] = (_Float16)0.f;
    }
  }
  float mlo = 0.f, mspan = 0.f;
  if (ncol < 3) { mlo = minmax[ncol]; mspan = minmax[3 + ncol] - mlo; }

#pragma unroll 1
  for (int tile = 0; tile < 2; ++tile) {
    const int rowPt = wave * 32 + tile * 16 + ncol;  // local point = A-frag row

    // A-frag from packed encodings: u[g] -> K pair j, contiguous K = 16h + s (same as B)
    AFrag a;
#pragma unroll
    for (int g = 0; g < 8; ++g) {
      int j = g + (hi ? 8 : 0);
      a.u[g] = encPack[rowPt][j];
    }

    v8f cz = {};
    // ---- layer 1: h1 = relu(enc @ W1^T) ----
    v8f d0 = wmma_f16(a.h, b1lo, cz);
    v8f d1 = wmma_f16(a.h, b1hi, cz);
#pragma unroll
    for (int v = 0; v < 8; ++v) {
      d0[v] = fmaxf(d0[v], 0.f);
      d1[v] = fmaxf(d1[v], 0.f);
    }
#pragma unroll
    for (int v = 0; v < 8; ++v) {           // D layout: M = v+mrow, N = ncol
      hbuf[wave][v + mrow][ncol]      = (_Float16)d0[v];
      hbuf[wave][v + mrow][ncol + 16] = (_Float16)d1[v];
    }
    __syncthreads();

    AFrag a2;
    {
      const unsigned* hr = reinterpret_cast<const unsigned*>(&hbuf[wave][ncol][0]);
#pragma unroll
      for (int g = 0; g < 8; ++g) {
        int j = g + (hi ? 8 : 0);
        a2.u[g] = hr[j];
      }
    }
    __syncthreads();

    // ---- layer 2: h2 = relu(h1 @ W2^T) ----
    v8f e0 = wmma_f16(a2.h, b2lo, cz);
    v8f e1 = wmma_f16(a2.h, b2hi, cz);
#pragma unroll
    for (int v = 0; v < 8; ++v) {
      e0[v] = fmaxf(e0[v], 0.f);
      e1[v] = fmaxf(e1[v], 0.f);
    }
#pragma unroll
    for (int v = 0; v < 8; ++v) {
      hbuf[wave][v + mrow][ncol]      = (_Float16)e0[v];
      hbuf[wave][v + mrow][ncol + 16] = (_Float16)e1[v];
    }
    __syncthreads();

    AFrag a3;
    {
      const unsigned* hr = reinterpret_cast<const unsigned*>(&hbuf[wave][ncol][0]);
#pragma unroll
      for (int g = 0; g < 8; ++g) {
        int j = g + (hi ? 8 : 0);
        a3.u[g] = hr[j];
      }
    }
    __syncthreads();

    // ---- layer 3: y = h2 @ W3^T (N=0..2 valid) + sigmoid + range map ----
    v8f y = wmma_f16(a3.h, b3, cz);
    if (ncol < 3) {
#pragma unroll
      for (int v = 0; v < 8; ++v) {
        int gp = blockBase + wave * 32 + tile * 16 + mrow + v;
        if (gp < npts) {
          float s = 1.f / (1.f + __expf(-y[v]));
          out[gp * 3 + ncol] = fmaf(s, mspan, mlo);
        }
      }
    }
  }
}

extern "C" void kernel_launch(void* const* d_in, const int* in_sizes, int n_in,
                              void* d_out, int out_size, void* d_ws, size_t ws_size,
                              hipStream_t stream) {
  (void)n_in; (void)out_size; (void)d_ws; (void)ws_size;
  const float* texc   = (const float*)d_in[0];
  const float* table  = (const float*)d_in[1];
  const float* W1     = (const float*)d_in[2];
  const float* W2     = (const float*)d_in[3];
  const float* W3     = (const float*)d_in[4];
  const float* aabb   = (const float*)d_in[5];
  const float* minmax = (const float*)d_in[6];
  float* out = (float*)d_out;

  int npts   = in_sizes[0] / 3;
  int blocks = (npts + 255) / 256;
  hipLaunchKernelGGL(mlptex3d_fused, dim3(blocks), dim3(256), 0, stream,
                     texc, table, W1, W2, W3, aabb, minmax, out, npts);
}
